// DownSample_90598040142151
// MI455X (gfx1250) — matrix operand order repaired and checked by reading, not compiled, not run on hardware
//
#include <hip/hip_runtime.h>
#include <hip/hip_bf16.h>

typedef __attribute__((ext_vector_type(16))) __bf16 v16bf;
typedef __attribute__((ext_vector_type(8)))  float  v8f;

// ---------------------------------------------------------------------------
// Problem constants
// ---------------------------------------------------------------------------
#define BATCH 16
#define CIN   256
#define HIN   64
#define WIN   64
#define HP    65          // padded-FIR intermediate height/width
#define WP    65
#define OC    256
#define OH    32
#define OW    32
#define KTOT  (9 * CIN)   // 2304, ordered k = tap*256 + ic, tap = kh*3+kw

// ---------------------------------------------------------------------------
// Kernel 1: depthwise 4x4 FIR (pad 2), f32 in NCHW -> bf16 intermediate NHWC.
// Compute with x-fastest thread mapping (coalesced f32 reads), transpose the
// 8x32 (x,c) tile through LDS, store with c-fastest mapping (coalesced bf16
// writes, 64B segments).
// ---------------------------------------------------------------------------
__global__ __launch_bounds__(256) void fir_pack_kernel(
    const float* __restrict__ x, __bf16* __restrict__ pad) {
  const int b  = blockIdx.z;          // 0..15
  const int y  = blockIdx.y;          // 0..64
  const int ct = blockIdx.x & 7;      // channel tile (32 channels)
  const int xt = blockIdx.x >> 3;     // x tile (8 columns), 9 tiles
  const int tid = threadIdx.x;

  __shared__ float tile[8 * 33];

  // compute mapping: x fastest
  const int cl = tid >> 3;            // 0..31
  const int xl = tid & 7;             // 0..7
  const int c  = ct * 32 + cl;
  const int xo = xt * 8 + xl;

  float acc = 0.0f;
  if (xo < WP) {
    const float f1[4] = {1.0f, 3.0f, 3.0f, 1.0f};
    const float* src = x + (((long)b * CIN + c) * HIN) * WIN;
#pragma unroll
    for (int i = 0; i < 4; ++i) {
      const int yi = y + i - 2;
      if (yi < 0 || yi >= HIN) continue;
#pragma unroll
      for (int j = 0; j < 4; ++j) {
        const int xi = xo + j - 2;
        if (xi < 0 || xi >= WIN) continue;
        acc += f1[i] * f1[j] * src[yi * WIN + xi];
      }
    }
    acc *= (1.0f / 64.0f);            // FIR normalization (sum = 64)
  }
  tile[xl * 33 + cl] = acc;
  __syncthreads();

  // store mapping: c fastest
  const int cs = tid & 31;
  const int xs0 = tid >> 5;           // 0..7
  const int xsg = xt * 8 + xs0;
  if (xsg < WP) {
    const float v = tile[xs0 * 33 + cs];
    pad[(((long)b * HP + y) * WP + xsg) * CIN + ct * 32 + cs] = (__bf16)v;
  }
}

// ---------------------------------------------------------------------------
// Kernel 2: pack weights into WMMA B-fragment order (bf16).
// Fragment (kk, nt): 32 lanes x 16 bf16, contiguous.  Per the CDNA5 B 32x16
// layout: lane column oc = nt*16 + (lane&15); lanes<16 hold Krel 0..7,16..23,
// lanes>=16 hold Krel 8..15,24..31.  k = tap*256 + ic, kk = tap*8+icg.
// ---------------------------------------------------------------------------
__global__ __launch_bounds__(256) void wpack_kernel(
    const float* __restrict__ w, __bf16* __restrict__ wp) {
  const int idx = blockIdx.x * 256 + threadIdx.x;   // 0 .. 589823
  if (idx >= KTOT * OC) return;
  const int j    = idx & 15;
  const int lane = (idx >> 4) & 31;
  const int nt   = (idx >> 9) & 15;
  const int kk   = idx >> 13;                       // 0..71
  const int krel = ((j >> 3) << 4) + ((lane >> 4) << 3) + (j & 7);  // table order: lanes>=16 -> +8
  const int oc   = (nt << 4) + (lane & 15);
  const int ic   = ((kk & 7) << 5) + krel;
  const int tap  = kk >> 3;
  const int kh   = tap / 3;
  const int kw   = tap % 3;
  wp[idx] = (__bf16)w[(((long)oc * CIN + ic) * 3 + kh) * 3 + kw];
}

// ---------------------------------------------------------------------------
// Kernel 3: implicit-GEMM conv via v_wmma_f32_16x16x32_bf16.
// Block = 256 threads = 8 waves (4 M-waves x 2 N-waves), block tile 128Mx128N.
// Wave tile 32M x 64N = 2x4 accumulators of v8f.
// K loop: 9 taps x 8 channel-groups; A fragments come straight from the NHWC
// bf16 intermediate as two b128 loads/lane (matches 16-bit A VGPR layout);
// B fragments are two b128 loads/lane from the pre-packed weights.
// Epilogue: LDS transpose (2 passes of 64x132 f32) -> coalesced NCHW stores
// with fused bias.
// ---------------------------------------------------------------------------
__global__ __launch_bounds__(256) void conv_wmma_kernel(
    const __bf16* __restrict__ pad, const __bf16* __restrict__ wp,
    const float* __restrict__ bias, float* __restrict__ out) {
  const int tid   = threadIdx.x;
  const int lane  = tid & 31;
  const int wid   = tid >> 5;        // 0..7
  const int waveM = wid & 3;         // 0..3
  const int waveN = wid >> 2;        // 0..1
  const int lrow  = lane & 15;
  const int lhalf = lane >> 4;       // 0/1 -> K-half of the fragment

  const int blockM = blockIdx.x;     // 0..127  (BM = 128, stays inside one b)
  const int blockN = blockIdx.y;     // 0..1    (BN = 128)

  // Per-lane spatial base for the two A fragments (mi = 0,1).
  int sbase[2];
#pragma unroll
  for (int mi = 0; mi < 2; ++mi) {
    const int m  = blockM * 128 + waveM * 32 + mi * 16 + lrow;
    const int b  = m >> 10;
    const int oh = (m >> 5) & 31;
    const int ow = m & 31;
    sbase[mi] = (((b * HP + oh * 2) * WP) + ow * 2) * CIN + lhalf * 8;
  }
  const int ntBase = blockN * 8 + waveN * 4;

  v8f acc[2][4];
#pragma unroll
  for (int mi = 0; mi < 2; ++mi)
#pragma unroll
    for (int ni = 0; ni < 4; ++ni)
      acc[mi][ni] = (v8f)(0.0f);

  for (int tap = 0; tap < 9; ++tap) {
    const int kh = tap / 3;
    const int kw = tap % 3;
    const int tapOff = (kh * WP + kw) * CIN;
    if (tap < 8) {  // speculative prefetch of next tap's A row
      const int nt2 = tap + 1;
      __builtin_prefetch(pad + sbase[0] + ((nt2 / 3) * WP + (nt2 % 3)) * CIN, 0, 0);
    }
#pragma unroll
    for (int icg = 0; icg < 8; ++icg) {
      const int kk = tap * 8 + icg;

      v16bf afrag[2];
#pragma unroll
      for (int mi = 0; mi < 2; ++mi) {
        const __bf16* p = pad + sbase[mi] + tapOff + icg * 32;
        ((uint4*)&afrag[mi])[0] = *(const uint4*)(p);        // Krel {0..7}+8h
        ((uint4*)&afrag[mi])[1] = *(const uint4*)(p + 16);   // Krel {16..23}+8h
      }
#pragma unroll
      for (int ni = 0; ni < 4; ++ni) {
        v16bf bfrag;
        const __bf16* q = wp + (((kk * 16) + ntBase + ni) * 32 + lane) * 16;
        ((uint4*)&bfrag)[0] = *(const uint4*)(q);
        ((uint4*)&bfrag)[1] = *(const uint4*)(q + 8);
#pragma unroll
        for (int mi = 0; mi < 2; ++mi) {
          acc[mi][ni] = __builtin_amdgcn_wmma_f32_16x16x32_bf16(
              false, afrag[mi], false, bfrag, (short)0, acc[mi][ni],
              false, false);
        }
      }
    }
  }

  // ---- epilogue: transpose through LDS, coalesced NCHW stores + bias ------
  __shared__ float lds[64 * 132];    // 33.8 KB, stride 132 avoids conflicts
  const int bb     = (blockM * 128) >> 10;
  const int ohBase = ((blockM * 128) >> 5) & 31;

#pragma unroll
  for (int pass = 0; pass < 2; ++pass) {
    if (waveN == pass) {
#pragma unroll
      for (int mi = 0; mi < 2; ++mi)
#pragma unroll
        for (int ni = 0; ni < 4; ++ni)
#pragma unroll
          for (int r = 0; r < 8; ++r) {
            const int mloc = waveM * 32 + mi * 16 + r + lhalf * 8;  // 0..127
            const int nloc = ni * 16 + lrow;                        // 0..63
            lds[nloc * 132 + mloc] = acc[mi][ni][r];
          }
    }
    __syncthreads();
    // 64 oc x 4 oh = 256 rows; 8 waves -> 32 rows/wave; 32 lanes = one ow row
#pragma unroll
    for (int rr = 0; rr < 32; ++rr) {
      const int row = wid * 32 + rr;       // 0..255
      const int nl  = row >> 2;            // 0..63
      const int ohl = row & 3;
      const int oc  = blockN * 128 + pass * 64 + nl;
      const float v = lds[nl * 132 + ohl * 32 + lane] + bias[oc];
      out[((((long)bb * OC + oc) * OH) + (ohBase + ohl)) * OW + lane] = v;
    }
    __syncthreads();
  }
}

// ---------------------------------------------------------------------------
// Host launcher
// ---------------------------------------------------------------------------
extern "C" void kernel_launch(void* const* d_in, const int* in_sizes, int n_in,
                              void* d_out, int out_size, void* d_ws, size_t ws_size,
                              hipStream_t stream) {
  (void)in_sizes; (void)n_in; (void)out_size; (void)ws_size;
  const float* x = (const float*)d_in[0];   // [16,256,64,64]
  const float* w = (const float*)d_in[1];   // [256,256,3,3]
  const float* b = (const float*)d_in[2];   // [256]
  float* out = (float*)d_out;               // [16,256,32,32]

  // Workspace partition: packed weights (1.2 MB) then NHWC bf16 pad (34.6 MB)
  __bf16* wp  = (__bf16*)d_ws;
  __bf16* pad = (__bf16*)((char*)d_ws + (2u << 20));

  // 1) FIR + NHWC bf16 pack: grid (ctile*xtile, y, b)
  fir_pack_kernel<<<dim3(8 * 9, HP, BATCH), 256, 0, stream>>>(x, pad);

  // 2) Weight repack into B-fragment order: 589824 elements
  wpack_kernel<<<dim3((KTOT * OC + 255) / 256), 256, 0, stream>>>(w, wp);

  // 3) Implicit GEMM: M=16384 -> 128 tiles of 128; N=256 -> 2 tiles of 128
  conv_wmma_kernel<<<dim3(128, 2), 256, 0, stream>>>(pad, wp, b, out);
}
